// LlamaAttention_71313636983294
// MI455X (gfx1250) — matrix operand order repaired and checked by reading, not compiled, not run on hardware
//
#include <hip/hip_runtime.h>
#include <hip/hip_bf16.h>
#include <math.h>

// Problem constants (LlamaAttention reference)
#define NH     16
#define NKV    4
#define DHEAD  128
#define SEQ    2048
#define DMODEL 2048
#define BATCH  2
#define MTOT   (BATCH * SEQ)     // 4096 rows of activations
#define MIN_V  (-1.0e15f)

// CDNA5 async global->LDS staging (tracked with ASYNCcnt)
#define USE_ASYNC_LDS 1

typedef __attribute__((ext_vector_type(16))) __bf16 v16bf;
typedef __attribute__((ext_vector_type(8)))  __bf16 v8bf;
typedef __attribute__((ext_vector_type(8)))  float  v8f;

union V16 {
    v16bf v;
    v8bf  h[2];
    unsigned short u[16];
};

static __device__ __forceinline__ unsigned short f2bf(float f) {
    unsigned u = __float_as_uint(f);
    u += 0x7fffu + ((u >> 16) & 1u);       // round-to-nearest-even
    return (unsigned short)(u >> 16);
}

static __device__ __forceinline__ v8f wmma_bf16(v16bf a, v16bf b, v8f c) {
    // v_wmma_f32_16x16x32_bf16  D = A*B + C
    return __builtin_amdgcn_wmma_f32_16x16x32_bf16(
        /*neg_a=*/false, a, /*neg_b=*/false, b,
        /*c_mod=*/(short)0, c, /*reuse_a=*/false, /*reuse_b=*/false);
}

// ---------------------------------------------------------------------------
// fp32 -> bf16 conversion (8 elements / thread: 32B in, 16B out)
// ---------------------------------------------------------------------------
__global__ __launch_bounds__(256)
void k_f32_to_bf16(const float* __restrict__ in, unsigned short* __restrict__ out, int n8) {
    int i = blockIdx.x * 256 + threadIdx.x;
    if (i < n8) {
        const float4* p = (const float4*)in + (size_t)i * 2;
        float4 a = p[0], b = p[1];
        union { unsigned short u[8]; uint4 q; } r;
        r.u[0] = f2bf(a.x); r.u[1] = f2bf(a.y); r.u[2] = f2bf(a.z); r.u[3] = f2bf(a.w);
        r.u[4] = f2bf(b.x); r.u[5] = f2bf(b.y); r.u[6] = f2bf(b.z); r.u[7] = f2bf(b.w);
        ((uint4*)out)[i] = r.q;
    }
}

// ---------------------------------------------------------------------------
// bf16 GEMM:  C[M,N] = X[M,K] * W[N,K]^T   (K = DMODEL = 2048)
// Workgroup: 256 threads (8 waves). Block tile 128(M) x 128(N).
// Wave w computes rows rbase+16w .. +15 over all 128 columns (8 accum tiles).
// W tile (128 cols x 32 K) staged into LDS each k-step via async-to-LDS ops.
// MODE: 0 = Q proj (+RoPE, bf16 out, N=2048)
//       1 = K proj (+RoPE, bf16 out, N=512)
//       2 = V proj (bf16 out, stored transposed [b][kv][d][l])
//       3 = O proj (fp32 out to d_out, N=2048)
// ---------------------------------------------------------------------------
template <int MODE>
__global__ __launch_bounds__(256)
void k_gemm(const unsigned short* __restrict__ X,
            const unsigned short* __restrict__ W,
            unsigned short* __restrict__ OutBF,
            float* __restrict__ OutF32)
{
    __shared__ unsigned short sB[128 * 32];

    const int wave  = threadIdx.x >> 5;
    const int lane  = threadIdx.x & 31;
    const int lm    = lane & 15;
    const int lh    = lane >> 4;
    const int rbase = blockIdx.x * 128;
    const int cbase = blockIdx.y * 128;
    const int rm    = rbase + wave * 16 + lm;   // A-fragment row for this lane

    v8f acc[8];
#pragma unroll
    for (int t = 0; t < 8; ++t) {
#pragma unroll
        for (int i = 0; i < 8; ++i) acc[t][i] = 0.0f;
    }

    // Per-thread staging slot: thread moves W[cbase+row][k0+16*half .. +15] (32 bytes)
    const int srow  = threadIdx.x & 127;
    const int shalf = threadIdx.x >> 7;

    for (int k0 = 0; k0 < DMODEL; k0 += 32) {
        __syncthreads();   // previous tile fully consumed before overwrite
        {
            const unsigned short* gsrc = W + (size_t)(cbase + srow) * DMODEL + k0 + 16 * shalf;
#if USE_ASYNC_LDS
            // INST_OFFSET is added to BOTH global and LDS addresses, so two
            // b128 async ops off one base cover the 32-byte slot.
            unsigned ldsoff = (unsigned)(unsigned long long)(&sB[srow * 32 + 16 * shalf]);
            asm volatile("global_load_async_to_lds_b128 %0, %1, off"
                         :: "v"(ldsoff), "v"(gsrc) : "memory");
            asm volatile("global_load_async_to_lds_b128 %0, %1, off offset:16"
                         :: "v"(ldsoff), "v"(gsrc) : "memory");
#else
            const uint4* src = (const uint4*)gsrc;
            uint4* dst = (uint4*)(&sB[srow * 32 + 16 * shalf]);
            dst[0] = src[0];
            dst[1] = src[1];
#endif
            if (k0 + 32 < DMODEL)
                __builtin_prefetch(W + (size_t)(cbase + srow) * DMODEL + k0 + 32 + 16 * shalf, 0, 1);
        }
#if USE_ASYNC_LDS
        asm volatile("s_wait_asynccnt 0x0" ::: "memory");
#endif
        __syncthreads();

        // A fragment: lane m=lm, K = 16*lh + e
        V16 af;
        af.h[0] = *(const v8bf*)(X + (size_t)rm * DMODEL + k0 + 16 * lh);
        af.h[1] = *(const v8bf*)(X + (size_t)rm * DMODEL + k0 + 16 * lh + 8);

#pragma unroll
        for (int t = 0; t < 8; ++t) {
            // B fragment: lane n=lm -> output col cbase+16t+lm; K = 16*lh + e
            V16 bf;
            const v8bf* bp = (const v8bf*)(&sB[(t * 16 + lm) * 32 + 16 * lh]);
            bf.h[0] = bp[0];
            bf.h[1] = bp[1];
            acc[t] = wmma_bf16(af.v, bf.v, acc[t]);
        }
    }

    // Epilogue.  C layout: vgpr r -> row r (lanes 0-15) / row r+8 (lanes 16-31), col = lm.
#pragma unroll
    for (int r = 0; r < 8; ++r) {
        const int mglob = rbase + wave * 16 + r + 8 * lh;
        const int b     = mglob >> 11;          // / SEQ
        const int pos   = mglob & (SEQ - 1);    // % SEQ

        if (MODE == 0 || MODE == 1) {
            const int head = blockIdx.y;        // 128-col block == one head
            const int nout = (MODE == 0) ? (NH * DHEAD) : (NKV * DHEAD);
#pragma unroll
            for (int t = 0; t < 4; ++t) {
                const int d1 = t * 16 + lm;     // 0..63
                float x1 = acc[t][r];           // x[d1]
                float x2 = acc[t + 4][r];       // x[d1+64]
                float fr = (float)pos * __powf(10000.0f, -((float)d1) / 64.0f);
                float cs = __cosf(fr), sn = __sinf(fr);
                float q1 = x1 * cs - x2 * sn;   // rope: left half
                float q2 = x2 * cs + x1 * sn;   // rope: right half
                size_t base = (size_t)mglob * nout + head * DHEAD;
                OutBF[base + d1]      = f2bf(q1);
                OutBF[base + d1 + 64] = f2bf(q2);
            }
        } else if (MODE == 2) {
            const int kvh = blockIdx.y;
#pragma unroll
            for (int t = 0; t < 8; ++t) {
                const int d = t * 16 + lm;
                // V^T layout: [b][kv][d][l]
                OutBF[((size_t)((b * NKV + kvh) * DHEAD + d)) * SEQ + pos] = f2bf(acc[t][r]);
            }
        } else {
#pragma unroll
            for (int t = 0; t < 8; ++t)
                OutF32[(size_t)mglob * DMODEL + cbase + t * 16 + lm] = acc[t][r];
        }
    }
}

// ---------------------------------------------------------------------------
// Flash attention: 8 waves per workgroup, 1 wave = 16 queries of one (b, h).
// Keys processed 32 at a time (two 16x16 score tiles), online softmax,
// probabilities re-laid C->A format through a per-wave 16x32 LDS tile,
// PV via WMMA against V^T.
// ---------------------------------------------------------------------------
__global__ __launch_bounds__(256)
void k_attn(const unsigned short* __restrict__ Q,
            const unsigned short* __restrict__ K,
            const unsigned short* __restrict__ Vt,
            const int* __restrict__ amask,
            unsigned short* __restrict__ O)
{
    __shared__ float sPall[8][16][32];

    const int wave  = threadIdx.x >> 5;
    const int lane  = threadIdx.x & 31;
    const int lm    = lane & 15;
    const int lh    = lane >> 4;
    const int qt    = blockIdx.x * 8 + wave;
    const int h     = blockIdx.y;
    const int b     = blockIdx.z;
    const int kv    = h >> 2;               // H/KV = 4
    const int qbase = qt * 16;
    const float scale = 0.08838834764831845f;   // 1/sqrt(128)
    float (*sP)[32] = sPall[wave];

    // Preload Q A-fragments (4 k-steps over DH=128)
    v16bf qf[4];
    {
        const size_t qrow = ((size_t)(b * SEQ + qbase + lm)) * DMODEL + h * DHEAD;
#pragma unroll
        for (int dt = 0; dt < 4; ++dt) {
            V16 a;
            a.h[0] = *(const v8bf*)(Q + qrow + dt * 32 + 16 * lh);
            a.h[1] = *(const v8bf*)(Q + qrow + dt * 32 + 16 * lh + 8);
            qf[dt] = a.v;
        }
    }

    v8f acc[8];
#pragma unroll
    for (int t = 0; t < 8; ++t) {
#pragma unroll
        for (int i = 0; i < 8; ++i) acc[t][i] = 0.0f;
    }
    float mrun[8], lrun[8];
#pragma unroll
    for (int r = 0; r < 8; ++r) { mrun[r] = -1.0e30f; lrun[r] = 0.0f; }

    for (int kb = 0; kb < qbase + 16; kb += 32) {
        const int am0 = amask[b * SEQ + kb + lm];
        const int am1 = amask[b * SEQ + kb + 16 + lm];

        // S = Q * K^T  (two 16x16 tiles: keys kb..kb+15, kb+16..kb+31)
        v8f s1, s2;
#pragma unroll
        for (int i = 0; i < 8; ++i) { s1[i] = 0.0f; s2[i] = 0.0f; }
#pragma unroll
        for (int dt = 0; dt < 4; ++dt) {
            V16 b1, b2;
            const size_t kr1 = ((size_t)(b * SEQ + kb + lm)) * (NKV * DHEAD) + kv * DHEAD + dt * 32;
            b1.h[0] = *(const v8bf*)(K + kr1 + 16 * lh);
            b1.h[1] = *(const v8bf*)(K + kr1 + 16 * lh + 8);
            s1 = wmma_bf16(qf[dt], b1.v, s1);
            const size_t kr2 = ((size_t)(b * SEQ + kb + 16 + lm)) * (NKV * DHEAD) + kv * DHEAD + dt * 32;
            b2.h[0] = *(const v8bf*)(K + kr2 + 16 * lh);
            b2.h[1] = *(const v8bf*)(K + kr2 + 16 * lh + 8);
            s2 = wmma_bf16(qf[dt], b2.v, s2);
        }

        // Mask + online softmax (row reductions across the 16-lane half)
#pragma unroll
        for (int r = 0; r < 8; ++r) {
            const int qpos = qbase + r + 8 * lh;
            float v1 = s1[r] * scale;
            float v2 = s2[r] * scale;
            if (am0 == 0 || (kb + lm)      > qpos) v1 = MIN_V;
            if (am1 == 0 || (kb + 16 + lm) > qpos) v2 = MIN_V;

            float mx = fmaxf(v1, v2);
#pragma unroll
            for (int off = 8; off >= 1; off >>= 1) mx = fmaxf(mx, __shfl_xor(mx, off, 32));
            float mnew = fmaxf(mrun[r], mx);
            float p1 = __expf(v1 - mnew);
            float p2 = __expf(v2 - mnew);
            float rs = p1 + p2;
#pragma unroll
            for (int off = 8; off >= 1; off >>= 1) rs += __shfl_xor(rs, off, 32);
            float corr = __expf(mrun[r] - mnew);
            lrun[r] = lrun[r] * corr + rs;
            mrun[r] = mnew;
#pragma unroll
            for (int t = 0; t < 8; ++t) acc[t][r] *= corr;

            // C-layout -> LDS (row = m, col = key index within block)
            sP[r + 8 * lh][lm]      = p1;
            sP[r + 8 * lh][lm + 16] = p2;
        }

        // Rebuild P as a 16x32 bf16 A-fragment (LDS in-order within wave)
        V16 pf;
        {
            const float* pr = &sP[lm][0];
#pragma unroll
            for (int e = 0; e < 8; ++e) {
                pf.u[e]     = f2bf(pr[16 * lh + e]);
                pf.u[8 + e] = f2bf(pr[16 * lh + 8 + e]);
            }
        }

        // acc += P * V   (B operand from V^T rows: lane n = d column)
#pragma unroll
        for (int t = 0; t < 8; ++t) {
            V16 bv;
            const size_t vr = ((size_t)((b * NKV + kv) * DHEAD + t * 16 + lm)) * SEQ + kb + 16 * lh;
            bv.h[0] = *(const v8bf*)(Vt + vr);
            bv.h[1] = *(const v8bf*)(Vt + vr + 8);
            acc[t] = wmma_bf16(pf.v, bv.v, acc[t]);
        }
    }

    // Normalize and write attention output (bf16, [b*L][H*DH])
#pragma unroll
    for (int r = 0; r < 8; ++r) {
        const float inv = 1.0f / lrun[r];
        const size_t orow = ((size_t)(b * SEQ + qbase + r + 8 * lh)) * DMODEL + h * DHEAD;
#pragma unroll
        for (int t = 0; t < 8; ++t)
            O[orow + t * 16 + lm] = f2bf(acc[t][r] * inv);
    }
}

// ---------------------------------------------------------------------------
// Host-side orchestration
// ---------------------------------------------------------------------------
extern "C" void kernel_launch(void* const* d_in, const int* in_sizes, int n_in,
                              void* d_out, int out_size, void* d_ws, size_t ws_size,
                              hipStream_t stream)
{
    (void)in_sizes; (void)n_in; (void)out_size; (void)ws_size;

    const float* hid = (const float*)d_in[0];
    const int*   am  = (const int*)d_in[1];
    const float* Wq  = (const float*)d_in[2];
    const float* Wk  = (const float*)d_in[3];
    const float* Wv  = (const float*)d_in[4];
    const float* Wo  = (const float*)d_in[5];
    float* out = (float*)d_out;

    char* ws = (char*)d_ws;
    unsigned short* Xbf  = (unsigned short*)(ws + 0);          // 16,777,216 B
    unsigned short* Wqbf = (unsigned short*)(ws + 16777216);   //  8,388,608 B
    unsigned short* Wkbf = (unsigned short*)(ws + 25165824);   //  2,097,152 B
    unsigned short* Wvbf = (unsigned short*)(ws + 27262976);   //  2,097,152 B
    unsigned short* Wobf = (unsigned short*)(ws + 29360128);   //  8,388,608 B
    unsigned short* Qbf  = (unsigned short*)(ws + 37748736);   // 16,777,216 B
    unsigned short* Kbf  = (unsigned short*)(ws + 54525952);   //  4,194,304 B
    unsigned short* Vtbf = (unsigned short*)(ws + 58720256);   //  4,194,304 B
    unsigned short* Abf  = (unsigned short*)(ws + 62914560);   // 16,777,216 B  (ends ~76 MB)

    int n8;
    n8 = (MTOT * DMODEL) / 8;
    k_f32_to_bf16<<<(n8 + 255) / 256, 256, 0, stream>>>(hid, Xbf, n8);
    n8 = (DMODEL * DMODEL) / 8;
    k_f32_to_bf16<<<(n8 + 255) / 256, 256, 0, stream>>>(Wq, Wqbf, n8);
    n8 = (NKV * DHEAD * DMODEL) / 8;
    k_f32_to_bf16<<<(n8 + 255) / 256, 256, 0, stream>>>(Wk, Wkbf, n8);
    k_f32_to_bf16<<<(n8 + 255) / 256, 256, 0, stream>>>(Wv, Wvbf, n8);
    n8 = (DMODEL * DMODEL) / 8;
    k_f32_to_bf16<<<(n8 + 255) / 256, 256, 0, stream>>>(Wo, Wobf, n8);

    // Projections (+RoPE for Q/K, transpose store for V)
    k_gemm<0><<<dim3(MTOT / 128, (NH * DHEAD) / 128), 256, 0, stream>>>(Xbf, Wqbf, Qbf, nullptr);
    k_gemm<1><<<dim3(MTOT / 128, (NKV * DHEAD) / 128), 256, 0, stream>>>(Xbf, Wkbf, Kbf, nullptr);
    k_gemm<2><<<dim3(MTOT / 128, (NKV * DHEAD) / 128), 256, 0, stream>>>(Xbf, Wvbf, Vtbf, nullptr);

    // Flash attention: 8 waves / workgroup, one 16-query tile per wave per (b, h)
    k_attn<<<dim3(SEQ / 128, NH, BATCH), 256, 0, stream>>>(Qbf, Kbf, Vtbf, am, Abf);

    // Output projection -> fp32 result
    k_gemm<3><<<dim3(MTOT / 128, DMODEL / 128), 256, 0, stream>>>(Abf, Wobf, nullptr, out);
}
